// GraphAttentionLayer_13013750906974
// MI455X (gfx1250) — matrix operand order repaired and checked by reading, not compiled, not run on hardware
//
#include <hip/hip_runtime.h>
#include <hip/hip_bf16.h>
#include <math.h>

// ---------------- CDNA5 WMMA types ----------------
typedef __attribute__((ext_vector_type(16))) __bf16 v16bf;
typedef __attribute__((ext_vector_type(8)))  float  v8f;

#define IN_DIM   256
#define HEADS    8
#define CDIM     32
#define DH_DIM   1024
#define LN_EPS   1e-5f
#define NEG_SLP  0.2f
#define TPB      256

union Frag {
  v16bf v;
  uint4 q[2];
};

__device__ __forceinline__ unsigned short f2bf(float f) {
  unsigned u = __float_as_uint(f);
  u += 0x7FFFu + ((u >> 16) & 1u);           // round-to-nearest-even
  return (unsigned short)(u >> 16);
}

// order-preserving float <-> uint encoding for atomicMax-based segment max
__device__ __forceinline__ unsigned fenc(float f) {
  int i = __float_as_int(f);
  return (i < 0) ? ~(unsigned)i : ((unsigned)i | 0x80000000u);
}
__device__ __forceinline__ float fdec(unsigned u) {
  int i = (u & 0x80000000u) ? (int)(u & 0x7FFFFFFFu) : ~(int)u;
  return __int_as_float(i);
}

__device__ __forceinline__ void blockReduceAdd2(float a, float b, float* o1, float* o2) {
  __shared__ float s1[TPB];
  __shared__ float s2[TPB];
  int t = threadIdx.x;
  s1[t] = a; s2[t] = b;
  __syncthreads();
  for (int o = TPB / 2; o > 0; o >>= 1) {
    if (t < o) { s1[t] += s1[t + o]; s2[t] += s2[t + o]; }
    __syncthreads();
  }
  if (t == 0) { atomicAdd(o1, s1[0]); atomicAdd(o2, s2[0]); }
}

// ---------------- init / conversion ----------------
__global__ void k_init(float* agg, float* denom, unsigned* mkey, float* scal,
                       size_t nAgg, int nNH) {
  size_t i = (size_t)blockIdx.x * blockDim.x + threadIdx.x;
  size_t stride = (size_t)gridDim.x * blockDim.x;
  for (size_t j = i; j < nAgg; j += stride) agg[j] = 0.f;
  for (size_t j = i; j < (size_t)nNH; j += stride) { denom[j] = 0.f; mkey[j] = 0u; }
  if (i < 4) scal[i] = 0.f;
}

__global__ void k_f32_to_bf16(const float* __restrict__ in, unsigned short* __restrict__ out,
                              size_t n) {
  size_t i = (size_t)blockIdx.x * blockDim.x + threadIdx.x;
  if (i < n) out[i] = f2bf(in[i]);
}

// in: [K][Nc] f32 row-major  ->  out: [Nc][K] bf16 (so B columns become contiguous)
__global__ void k_transpose_bf16(const float* __restrict__ in, unsigned short* __restrict__ out,
                                 int K, int Nc) {
  size_t i = (size_t)blockIdx.x * blockDim.x + threadIdx.x;
  if (i >= (size_t)K * Nc) return;
  int n = (int)(i / K);
  int k = (int)(i % K);
  out[i] = f2bf(in[(size_t)k * Nc + n]);
}

// ---------------- GEMM 1: xl = x@Wl+bl, xr = x@Wr+br (fused, shared A) ----------------
__global__ __launch_bounds__(TPB) void k_gemm_gat(
    const unsigned short* __restrict__ xbf,
    const unsigned short* __restrict__ wlT, const unsigned short* __restrict__ wrT,
    const float* __restrict__ bl, const float* __restrict__ br,
    float* __restrict__ xl, float* __restrict__ xr, int nRowTiles) {
  const int lane = threadIdx.x & 31;
  const int wave = (blockIdx.x << 3) + (threadIdx.x >> 5);
  const int tileM = wave >> 4;            // 256/16 = 16 column tiles
  const int tileN = wave & 15;
  if (tileM >= nRowTiles) return;         // wave-uniform
  const int ar = tileM * 16 + (lane & 15);
  const int bn = tileN * 16 + (lane & 15);
  const int aoff = (lane >> 4) * 16;
  const int boff = (lane >> 4) * 16;
  const unsigned short* aRow  = xbf + (size_t)ar * IN_DIM;
  const unsigned short* blRow = wlT + (size_t)bn * IN_DIM;
  const unsigned short* brRow = wrT + (size_t)bn * IN_DIM;
  v8f accL = {}; v8f accR = {};
#pragma unroll
  for (int kk = 0; kk < IN_DIM; kk += 32) {
    Frag a, b0, b1;
    a.q[0]  = *(const uint4*)(aRow  + kk + aoff);
    a.q[1]  = *(const uint4*)(aRow  + kk + aoff + 8);
    b0.q[0] = *(const uint4*)(blRow + kk + boff);
    b0.q[1] = *(const uint4*)(blRow + kk + boff + 8);
    b1.q[0] = *(const uint4*)(brRow + kk + boff);
    b1.q[1] = *(const uint4*)(brRow + kk + boff + 8);
    accL = __builtin_amdgcn_wmma_f32_16x16x32_bf16(false, a.v, false, b0.v, (short)0, accL, false, false);
    accR = __builtin_amdgcn_wmma_f32_16x16x32_bf16(false, a.v, false, b1.v, (short)0, accR, false, false);
  }
  const int col = lane & 15;
  const int rb  = (lane >> 4) * 8;
  const float vbl = bl[tileN * 16 + col];
  const float vbr = br[tileN * 16 + col];
#pragma unroll
  for (int r = 0; r < 8; ++r) {
    size_t idx = (size_t)(tileM * 16 + rb + r) * IN_DIM + tileN * 16 + col;
    xl[idx] = accL[r] + vbl;
    xr[idx] = accR[r] + vbr;
  }
}

// ---------------- edge phase ----------------
__global__ void k_logits(const float* __restrict__ xl, const float* __restrict__ xr,
                         const int* __restrict__ ei, const float* __restrict__ att,
                         float* __restrict__ plog, unsigned* __restrict__ mkey,
                         int E, int N) {
  int tid = blockIdx.x * blockDim.x + threadIdx.x;
  int total = (E + N) * HEADS;
  if (tid >= total) return;
  int e = tid >> 3, h = tid & 7;
  int src, dst;
  if (e < E) { src = ei[e]; dst = ei[E + e]; } else { src = dst = e - E; }
  const float* pl = xl + (size_t)src * IN_DIM + h * CDIM;
  const float* pr = xr + (size_t)dst * IN_DIM + h * CDIM;
  const float* pa = att + h * CDIM;
  float acc = 0.f;
#pragma unroll
  for (int c = 0; c < CDIM; ++c) {
    float a = pl[c] + pr[c];
    a = (a > 0.f) ? a : NEG_SLP * a;
    acc += a * pa[c];
  }
  plog[tid] = acc;
  atomicMax(&mkey[dst * HEADS + h], fenc(acc));
}

__global__ void k_softmax_p(float* __restrict__ plog, const unsigned* __restrict__ mkey,
                            float* __restrict__ denom, const int* __restrict__ ei,
                            int E, int N) {
  int tid = blockIdx.x * blockDim.x + threadIdx.x;
  int total = (E + N) * HEADS;
  if (tid >= total) return;
  int e = tid >> 3, h = tid & 7;
  int dst = (e < E) ? ei[E + e] : (e - E);
  float p = expf(plog[tid] - fdec(mkey[dst * HEADS + h]));
  plog[tid] = p;
  atomicAdd(&denom[dst * HEADS + h], p);
}

__global__ void k_scatter(const float* __restrict__ plog, const float* __restrict__ denom,
                          const float* __restrict__ xl, const int* __restrict__ ei,
                          float* __restrict__ agg, int E, int N) {
  int tid = blockIdx.x * blockDim.x + threadIdx.x;
  int total = (E + N) * HEADS;
  if (tid >= total) return;
  int e = tid >> 3, h = tid & 7;
  int src, dst;
  if (e < E) { src = ei[e]; dst = ei[E + e]; } else { src = dst = e - E; }
  float alpha = plog[tid] / denom[dst * HEADS + h];
  const float* pl = xl + (size_t)src * IN_DIM + h * CDIM;
  float* pd = agg + (size_t)dst * IN_DIM + h * CDIM;
#pragma unroll
  for (int c = 0; c < CDIM; ++c) atomicAdd(&pd[c], alpha * pl[c]);
}

// ---------------- residual + graph-LN stats ----------------
__global__ __launch_bounds__(TPB) void k_resid_stats(
    const float* __restrict__ x, const float* __restrict__ agg,
    const float* __restrict__ bgat, float* __restrict__ y1,
    float* __restrict__ scal, size_t total) {
  size_t i = (size_t)blockIdx.x * TPB + threadIdx.x;
  float v = 0.f;
  if (i < total) {
    v = x[i] + agg[i] + bgat[i & (IN_DIM - 1)];
    y1[i] = v;
  }
  blockReduceAdd2(v, v * v, &scal[0], &scal[1]);
}

// graph layernorm apply; writes bf16 (for next GEMM) or f32 (final output)
__global__ void k_layernorm(const float* __restrict__ v, const float* __restrict__ scal, int si,
                            const float* __restrict__ g, const float* __restrict__ be,
                            float* __restrict__ outF, unsigned short* __restrict__ outBf,
                            size_t total, float invCnt) {
  size_t i = (size_t)blockIdx.x * blockDim.x + threadIdx.x;
  if (i >= total) return;
  float mu  = scal[si] * invCnt;
  float var = scal[si + 1] * invCnt - mu * mu;
  float inv = 1.f / (sqrtf(fmaxf(var, 0.f)) + LN_EPS);
  float r = (v[i] - mu) * inv * g[i & (IN_DIM - 1)] + be[i & (IN_DIM - 1)];
  if (outBf) outBf[i] = f2bf(r);
  else       outF[i]  = r;
}

// ---------------- GEMM 2: h2 = relu(h1@W2+b2), bf16 out ----------------
__global__ __launch_bounds__(TPB) void k_gemm_mlp1(
    const unsigned short* __restrict__ h1bf, const unsigned short* __restrict__ w2T,
    const float* __restrict__ b2, unsigned short* __restrict__ h2bf, int nRowTiles) {
  const int lane = threadIdx.x & 31;
  const int wave = (blockIdx.x << 3) + (threadIdx.x >> 5);
  const int tileM = wave >> 6;            // 1024/16 = 64 column tiles
  const int tileN = wave & 63;
  if (tileM >= nRowTiles) return;
  const int ar = tileM * 16 + (lane & 15);
  const int bn = tileN * 16 + (lane & 15);
  const int aoff = (lane >> 4) * 16;
  const int boff = (lane >> 4) * 16;
  const unsigned short* aRow = h1bf + (size_t)ar * IN_DIM;
  const unsigned short* bRow = w2T  + (size_t)bn * IN_DIM;
  v8f acc = {};
#pragma unroll
  for (int kk = 0; kk < IN_DIM; kk += 32) {
    Frag a, b0;
    a.q[0]  = *(const uint4*)(aRow + kk + aoff);
    a.q[1]  = *(const uint4*)(aRow + kk + aoff + 8);
    b0.q[0] = *(const uint4*)(bRow + kk + boff);
    b0.q[1] = *(const uint4*)(bRow + kk + boff + 8);
    acc = __builtin_amdgcn_wmma_f32_16x16x32_bf16(false, a.v, false, b0.v, (short)0, acc, false, false);
  }
  const int col = lane & 15;
  const int rb  = (lane >> 4) * 8;
  const float bias = b2[tileN * 16 + col];
#pragma unroll
  for (int r = 0; r < 8; ++r) {
    float v = fmaxf(acc[r] + bias, 0.f);
    h2bf[(size_t)(tileM * 16 + rb + r) * DH_DIM + tileN * 16 + col] = f2bf(v);
  }
}

// ---------------- GEMM 3: z = h2@W3+b3, fused LN3 stats ----------------
__global__ __launch_bounds__(TPB) void k_gemm_mlp2(
    const unsigned short* __restrict__ h2bf, const unsigned short* __restrict__ w3T,
    const float* __restrict__ b3, float* __restrict__ z,
    float* __restrict__ scal, int nRowTiles) {
  const int lane = threadIdx.x & 31;
  const int wave = (blockIdx.x << 3) + (threadIdx.x >> 5);
  const int tileM = wave >> 4;            // 256/16 = 16 column tiles
  const int tileN = wave & 15;
  const bool active = (tileM < nRowTiles);
  float ls = 0.f, lq = 0.f;
  if (active) {
    const int ar = tileM * 16 + (lane & 15);
    const int bn = tileN * 16 + (lane & 15);
    const int aoff = (lane >> 4) * 16;
    const int boff = (lane >> 4) * 16;
    const unsigned short* aRow = h2bf + (size_t)ar * DH_DIM;
    const unsigned short* bRow = w3T  + (size_t)bn * DH_DIM;
    v8f acc = {};
#pragma unroll 4
    for (int kk = 0; kk < DH_DIM; kk += 32) {
      Frag a, b0;
      a.q[0]  = *(const uint4*)(aRow + kk + aoff);
      a.q[1]  = *(const uint4*)(aRow + kk + aoff + 8);
      b0.q[0] = *(const uint4*)(bRow + kk + boff);
      b0.q[1] = *(const uint4*)(bRow + kk + boff + 8);
      acc = __builtin_amdgcn_wmma_f32_16x16x32_bf16(false, a.v, false, b0.v, (short)0, acc, false, false);
    }
    const int col = lane & 15;
    const int rb  = (lane >> 4) * 8;
    const float bias = b3[tileN * 16 + col];
#pragma unroll
    for (int r = 0; r < 8; ++r) {
      float v = acc[r] + bias;
      z[(size_t)(tileM * 16 + rb + r) * IN_DIM + tileN * 16 + col] = v;
      ls += v;
      lq += v * v;
    }
  }
  blockReduceAdd2(ls, lq, &scal[2], &scal[3]);
}

// ---------------- host-side orchestration ----------------
extern "C" void kernel_launch(void* const* d_in, const int* in_sizes, int n_in,
                              void* d_out, int out_size, void* d_ws, size_t ws_size,
                              hipStream_t stream) {
  const float* x    = (const float*)d_in[0];
  const int*   ei   = (const int*)d_in[1];
  const float* Wl   = (const float*)d_in[2];
  const float* bl   = (const float*)d_in[3];
  const float* Wr   = (const float*)d_in[4];
  const float* br   = (const float*)d_in[5];
  const float* att  = (const float*)d_in[6];
  const float* bgat = (const float*)d_in[7];
  const float* g1   = (const float*)d_in[8];
  const float* be1  = (const float*)d_in[9];
  const float* W2   = (const float*)d_in[10];
  const float* b2   = (const float*)d_in[11];
  const float* W3   = (const float*)d_in[12];
  const float* b3   = (const float*)d_in[13];
  const float* g3   = (const float*)d_in[14];
  const float* be3  = (const float*)d_in[15];

  const int N = in_sizes[0] / IN_DIM;
  const int E = in_sizes[1] / 2;
  const size_t nx = (size_t)N * IN_DIM;

  // workspace carve-out (256B aligned)
  size_t off = 0;
  auto alloc = [&](size_t bytes) -> char* {
    char* p = (char*)d_ws + off;
    off += (bytes + 255) & ~(size_t)255;
    return p;
  };
  unsigned short* xbf  = (unsigned short*)alloc(nx * 2);                        // aliased: h1 bf16
  unsigned short* wlT  = (unsigned short*)alloc((size_t)IN_DIM * IN_DIM * 2);
  unsigned short* wrT  = (unsigned short*)alloc((size_t)IN_DIM * IN_DIM * 2);
  unsigned short* w2T  = (unsigned short*)alloc((size_t)IN_DIM * DH_DIM * 2);   // [1024][256]
  unsigned short* w3T  = (unsigned short*)alloc((size_t)DH_DIM * IN_DIM * 2);   // [256][1024]
  float*          xl   = (float*)alloc(nx * 4);                                 // aliased: z
  float*          xr   = (float*)alloc(nx * 4);                                 // aliased: y1
  float*          plog = (float*)alloc((size_t)(E + N) * HEADS * 4);
  unsigned*       mkey = (unsigned*)alloc((size_t)N * HEADS * 4);
  float*          den  = (float*)alloc((size_t)N * HEADS * 4);
  float*          agg  = (float*)alloc(nx * 4);
  unsigned short* h2bf = (unsigned short*)alloc((size_t)N * DH_DIM * 2);
  float*          scal = (float*)alloc(256);
  unsigned short* h1bf = xbf;   // x_bf16 dead after GEMM1
  float*          y1   = xr;    // xr dead after logits
  float*          z    = xl;    // xl dead after scatter

  const int rowTiles = (N + 15) / 16;
  const int EL = E + N;
  const size_t nEH = (size_t)EL * HEADS;

  k_init<<<2048, TPB, 0, stream>>>(agg, den, mkey, scal, nx, N * HEADS);

  k_f32_to_bf16<<<(unsigned)((nx + TPB - 1) / TPB), TPB, 0, stream>>>(x, xbf, nx);
  k_transpose_bf16<<<(IN_DIM * IN_DIM + TPB - 1) / TPB, TPB, 0, stream>>>(Wl, wlT, IN_DIM, IN_DIM);
  k_transpose_bf16<<<(IN_DIM * IN_DIM + TPB - 1) / TPB, TPB, 0, stream>>>(Wr, wrT, IN_DIM, IN_DIM);
  k_transpose_bf16<<<(IN_DIM * DH_DIM + TPB - 1) / TPB, TPB, 0, stream>>>(W2, w2T, IN_DIM, DH_DIM);
  k_transpose_bf16<<<(DH_DIM * IN_DIM + TPB - 1) / TPB, TPB, 0, stream>>>(W3, w3T, DH_DIM, IN_DIM);

  // GATv2 linear transforms (fused xl/xr, bf16 WMMA)
  k_gemm_gat<<<(rowTiles * 16 + 7) / 8, TPB, 0, stream>>>(xbf, wlT, wrT, bl, br, xl, xr, rowTiles);

  // edge attention: logits + segment-max, softmax numerator + segment-sum, scatter
  unsigned gEH = (unsigned)((nEH + TPB - 1) / TPB);
  k_logits   <<<gEH, TPB, 0, stream>>>(xl, xr, ei, att, plog, mkey, E, N);
  k_softmax_p<<<gEH, TPB, 0, stream>>>(plog, mkey, den, ei, E, N);
  k_scatter  <<<gEH, TPB, 0, stream>>>(plog, den, xl, ei, agg, E, N);

  // residual + graph-LN1 (global mean/var) -> h1 in bf16
  k_resid_stats<<<(unsigned)((nx + TPB - 1) / TPB), TPB, 0, stream>>>(x, agg, bgat, y1, scal, nx);
  k_layernorm<<<(unsigned)((nx + TPB - 1) / TPB), TPB, 0, stream>>>(
      y1, scal, 0, g1, be1, (float*)nullptr, h1bf, nx, 1.f / (float)nx);

  // MLP
  k_gemm_mlp1<<<(rowTiles * 64 + 7) / 8, TPB, 0, stream>>>(h1bf, w2T, b2, h2bf, rowTiles);
  k_gemm_mlp2<<<(rowTiles * 16 + 7) / 8, TPB, 0, stream>>>(h2bf, w3T, b3, z, scal, rowTiles);

  // graph-LN3 -> output (f32)
  k_layernorm<<<(unsigned)((nx + TPB - 1) / TPB), TPB, 0, stream>>>(
      z, scal, 2, g3, be3, (float*)d_out, (unsigned short*)nullptr, nx, 1.f / (float)nx);

  (void)n_in; (void)out_size; (void)ws_size;
}
